// SingleBNN_55061480735466
// MI455X (gfx1250) — 8-bit matrix operand order repaired and checked by reading, compiled, not run on hardware
//
#include <hip/hip_runtime.h>
#include <hip/hip_bf16.h>
#include <stdint.h>
#include <math.h>

typedef __attribute__((ext_vector_type(8))) int v8i;
typedef __attribute__((ext_vector_type(4))) int v4i;

typedef __attribute__((address_space(1))) v4i as1_v4i;
typedef __attribute__((address_space(3))) v4i as3_v4i;

#define BNN_EPS 1e-5f

#if __has_builtin(__builtin_amdgcn_global_load_async_to_lds_b128) && \
    __has_builtin(__builtin_amdgcn_s_wait_asynccnt)
#define USE_ASYNC_LDS 1
#else
#define USE_ASYNC_LDS 0
#endif

__global__ __launch_bounds__(256)
void prepack_kernel(const float* __restrict__ wb, const float* __restrict__ bb,
                    const float* __restrict__ w9,
                    const float* __restrict__ g16, const float* __restrict__ be16,
                    const float* __restrict__ m16, const float* __restrict__ v16,
                    int* __restrict__ bw, float* __restrict__ thr,
                    int8_t* __restrict__ w9p)
{
    const int L   = blockIdx.x;
    const int tid = threadIdx.x;
    if (L < 8) {
        for (int d = tid; d < 768; d += 256) {
            const int s    = d >> 8;
            const int rem  = d & 255;
            const int lane = rem >> 3;
            const int v    = rem & 7;
            const int half = lane >> 4;
            const int n    = lane & 15;
            const int kbase = ((v < 4) ? (v << 2) : (32 + ((v - 4) << 2))) + (half << 4);
            unsigned dw = 0u;
#pragma unroll
            for (int j = 0; j < 4; ++j) {
                const int kg = s * 64 + kbase + j;
                unsigned byte = 0u;
                if (kg < 144) {
                    const int tap = kg >> 4;
                    const int c   = kg & 15;
                    const float wv = wb[(((size_t)L * 16 + n) * 16 + c) * 9 + tap];
                    byte = (wv >= 0.f) ? 0x01u : 0xFFu;
                }
                dw |= (byte & 0xFFu) << (8 * j);
            }
            bw[L * 768 + d] = (int)dw;
        }
        if (tid < 16) {
            const int n = tid;
            const float iv = g16[(L + 1) * 16 + n] * rsqrtf(v16[(L + 1) * 16 + n] + BNN_EPS);
            thr[L * 16 + n] = bb[L * 16 + n] +
                              (be16[(L + 1) * 16 + n] - m16[(L + 1) * 16 + n] * iv) / iv;
        }
    } else {
        for (int i = tid; i < 10 * 144; i += 256) {
            const int cls = i / 144;
            const int r   = i % 144;
            const int tap = r >> 4;
            const int c   = r & 15;
            const float wv = w9[((size_t)cls * 16 + c) * 9 + tap];
            w9p[i] = (wv >= 0.f) ? (int8_t)1 : (int8_t)-1;
        }
    }
}

__global__ __launch_bounds__(256)
void conv0_kernel(const float* __restrict__ x, const float* __restrict__ w0,
                  const float* __restrict__ b0,
                  const float* __restrict__ g16, const float* __restrict__ be16,
                  const float* __restrict__ m16, const float* __restrict__ v16,
                  int8_t* __restrict__ out)
{
    __shared__ float w[16 * 27];
    __shared__ float bias[16], inv[16], sh[16];
    const int tid = threadIdx.x;
    for (int i = tid; i < 432; i += 256) w[i] = w0[i];
    if (tid < 16) {
        bias[tid] = b0[tid];
        const float iv = g16[tid] * rsqrtf(v16[tid] + BNN_EPS);
        inv[tid] = iv;
        sh[tid]  = be16[tid] - m16[tid] * iv;
    }
    __syncthreads();

    const size_t gid = (size_t)blockIdx.x * 256 + tid;
    const int b   = (int)(gid >> 12);
    const int pix = (int)(gid & 4095);
    const int oy  = pix >> 6, ox = pix & 63;

    float xin[27];
#pragma unroll
    for (int ic = 0; ic < 3; ++ic)
#pragma unroll
        for (int dy = 0; dy < 3; ++dy)
#pragma unroll
            for (int dx = 0; dx < 3; ++dx) {
                const int iy = oy + dy - 1, ix = ox + dx - 1;
                float val = 0.f;
                if (iy >= 0 && iy < 64 && ix >= 0 && ix < 64)
                    val = x[(((size_t)b * 3 + ic) * 64 + iy) * 64 + ix];
                xin[ic * 9 + dy * 3 + dx] = val;
            }

    union { int8_t c[16]; v4i v; } res;
#pragma unroll
    for (int oc = 0; oc < 16; ++oc) {
        float acc = bias[oc];
#pragma unroll
        for (int k = 0; k < 27; ++k) acc = fmaf(w[oc * 27 + k], xin[k], acc);
        const float y = acc * inv[oc] + sh[oc];
        res.c[oc] = (y >= 0.f) ? (int8_t)1 : (int8_t)-1;
    }
    *(v4i*)(out + gid * 16) = res.v;
}

__global__ __launch_bounds__(32)
void bconv_wmma(const int8_t* __restrict__ in, int8_t* __restrict__ out,
                int H, int W, int tpw_shift,
                const int* __restrict__ bfrag,
                const float* __restrict__ thresh)
{
    __shared__ __align__(16) int halo[400];

    const int lane = threadIdx.x & 31;
    const int tpw  = 1 << tpw_shift;
    const int b    = blockIdx.x >> (2 * tpw_shift);
    const int t    = blockIdx.x & (tpw * tpw - 1);
    const int ty8  = (t >> tpw_shift) << 3;
    const int tx8  = (t & (tpw - 1)) << 3;

    __builtin_prefetch(bfrag, 0, 3);

#pragma unroll
    for (int r = 0; r < 4; ++r) {
        const int p = lane + r * 32;
        if (p < 100) {
            const int py = p / 10, px = p % 10;
            const int iy = ty8 + py - 1, ix = tx8 + px - 1;
            if (iy >= 0 && iy < H && ix >= 0 && ix < W) {
#if USE_ASYNC_LDS
                __builtin_amdgcn_global_load_async_to_lds_b128(
                    (as1_v4i*)(uintptr_t)(in + (((size_t)b * H + iy) * W + ix) * 16),
                    (as3_v4i*)(v4i*)(&halo[p * 4]), 0, 0);
#else
                *(v4i*)(&halo[p * 4]) =
                    *(const v4i*)(in + (((size_t)b * H + iy) * W + ix) * 16);
#endif
            } else {
                const v4i z = {0, 0, 0, 0};
                *(v4i*)(&halo[p * 4]) = z;
            }
        }
    }
#if USE_ASYNC_LDS
    __builtin_amdgcn_s_wait_asynccnt(0);
#endif
    __syncthreads();

    v8i bf[3];
#pragma unroll
    for (int s = 0; s < 3; ++s) {
        const v4i* p = (const v4i*)(bfrag + s * 256 + lane * 8);
        union { v8i v; v4i h[2]; } u;
        u.h[0] = p[0];
        u.h[1] = p[1];
        bf[s] = u.v;
    }

    const int half = lane >> 4;
    const int m    = lane & 15;
    const int my   = m >> 2, mx = m & 3;
    const float T  = thresh[m];

#pragma unroll
    for (int q = 0; q < 4; ++q) {
        const int qy = (q >> 1) << 2;
        const int qx = (q & 1) << 2;

        v8i acc = {0, 0, 0, 0, 0, 0, 0, 0};
#pragma unroll
        for (int s = 0; s < 3; ++s) {
            v8i a;
#pragma unroll
            for (int v = 0; v < 8; ++v) {
                const int klocal = ((v >> 2) << 5) + ((v & 3) << 2) + (half << 4);
                const int kg = s * 64 + klocal;
                int val = 0;
                if (kg < 144) {
                    const int tap = kg >> 4;
                    const int c   = kg & 15;
                    const int ty  = tap / 3, tx = tap % 3;
                    const int py  = qy + my + ty;
                    const int px  = qx + mx + tx;
                    val = halo[(((py * 10 + px) << 4) + c) >> 2];
                }
                a[v] = val;
            }
            acc = __builtin_amdgcn_wmma_i32_16x16x64_iu8(
                       true, a,  true, bf[s], acc,
                       false,  false);
        }

#pragma unroll
        for (int v = 0; v < 8; ++v) {
            const int M  = v + (half << 3);
            const int oy = ty8 + qy + (M >> 2);
            const int ox = tx8 + qx + (M & 3);
            const float y = (float)acc[v] + T;
            out[(((size_t)b * H + oy) * W + ox) * 16 + m] =
                (y >= 0.f) ? (int8_t)1 : (int8_t)-1;
        }
    }
}

__global__ __launch_bounds__(256)
void pool_kernel(const int8_t* __restrict__ in, int8_t* __restrict__ out,
                 int wo_shift)
{
    const int gid = blockIdx.x * 256 + threadIdx.x;
    const int b   = gid >> (2 * wo_shift);
    const int p   = gid & ((1 << (2 * wo_shift)) - 1);
    const int oy  = p >> wo_shift;
    const int ox  = p & ((1 << wo_shift) - 1);
    const int Wi  = 2 << wo_shift;
    const int Hi  = Wi;
    const int8_t* p00 = in + (((size_t)b * Hi + oy * 2) * Wi + ox * 2) * 16;
    const int8_t* p10 = p00 + (size_t)Wi * 16;
    union { int8_t c[16]; v4i v; } r;
#pragma unroll
    for (int c = 0; c < 16; ++c) {
        const int8_t m0 = p00[c] > p00[16 + c] ? p00[c] : p00[16 + c];
        const int8_t m1 = p10[c] > p10[16 + c] ? p10[c] : p10[16 + c];
        r.c[c] = m0 > m1 ? m0 : m1;
    }
    *(v4i*)(out + (size_t)gid * 16) = r.v;
}

__global__ __launch_bounds__(64)
void head_kernel(const int8_t* __restrict__ in,
                 const int8_t* __restrict__ w9p,
                 const float* __restrict__ b9,
                 const float* __restrict__ g10, const float* __restrict__ be10,
                 const float* __restrict__ m10, const float* __restrict__ v10,
                 float* __restrict__ out)
{
    __shared__ float red[10][64];
    __shared__ float logits[10];
    const int b  = blockIdx.x;
    const int t  = threadIdx.x;
    const int oy = t >> 3, ox = t & 7;

    union { int8_t c[9][16]; v4i v[9]; } xin;
#pragma unroll
    for (int dy = 0; dy < 3; ++dy)
#pragma unroll
        for (int dx = 0; dx < 3; ++dx) {
            const int iy = oy + dy - 1, ix = ox + dx - 1;
            v4i pix = {0, 0, 0, 0};
            if (iy >= 0 && iy < 8 && ix >= 0 && ix < 8)
                pix = *(const v4i*)(in + (((size_t)b * 8 + iy) * 8 + ix) * 16);
            xin.v[dy * 3 + dx] = pix;
        }

#pragma unroll
    for (int cls = 0; cls < 10; ++cls) {
        int sum = 0;
#pragma unroll
        for (int tap = 0; tap < 9; ++tap)
            for (int c = 0; c < 16; ++c)
                sum += (int)xin.c[tap][c] * (int)w9p[(cls * 9 + tap) * 16 + c];
        const float iv = g10[cls] * rsqrtf(v10[cls] + BNN_EPS);
        red[cls][t] = ((float)sum + b9[cls]) * iv + (be10[cls] - m10[cls] * iv);
    }
    __syncthreads();

    if (t < 10) {
        float s = 0.f;
        for (int i = 0; i < 64; ++i) s += red[t][i];
        logits[t] = s * (1.f / 64.f);
    }
    __syncthreads();
    if (t < 10) {
        float mx = logits[0];
        for (int i = 1; i < 10; ++i) mx = fmaxf(mx, logits[i]);
        float se = 0.f;
        for (int i = 0; i < 10; ++i) se += __expf(logits[i] - mx);
        out[(size_t)b * 10 + t] = logits[t] - mx - __logf(se);
    }
}

extern "C" void kernel_launch(void* const* d_in, const int* in_sizes, int n_in,
                              void* d_out, int out_size, void* d_ws, size_t ws_size,
                              hipStream_t stream)
{
    (void)n_in; (void)out_size; (void)ws_size;
    const float* x    = (const float*)d_in[0];
    const float* w0   = (const float*)d_in[1];
    const float* b0   = (const float*)d_in[2];
    const float* wb   = (const float*)d_in[3];
    const float* bb   = (const float*)d_in[4];
    const float* w9   = (const float*)d_in[5];
    const float* b9   = (const float*)d_in[6];
    const float* g16  = (const float*)d_in[7];
    const float* be16 = (const float*)d_in[8];
    const float* m16  = (const float*)d_in[9];
    const float* v16  = (const float*)d_in[10];
    const float* g10  = (const float*)d_in[11];
    const float* be10 = (const float*)d_in[12];
    const float* m10  = (const float*)d_in[13];
    const float* v10  = (const float*)d_in[14];

    const int B = in_sizes[0] / (3 * 64 * 64);

    int8_t* act0 = (int8_t*)d_ws;
    int8_t* act1 = act0 + (size_t)B * 64 * 64 * 16;
    int*    bw   = (int*)(act1 + (size_t)B * 64 * 64 * 16);
    float*  thr  = (float*)(bw + 8 * 768);
    int8_t* w9p  = (int8_t*)(thr + 8 * 16);

    prepack_kernel<<<9, 256, 0, stream>>>(wb, bb, w9, g16, be16, m16, v16,
                                          bw, thr, w9p);
    conv0_kernel<<<(B * 4096) / 256, 256, 0, stream>>>(x, w0, b0,
                                                       g16, be16, m16, v16, act0);

    auto bconv = [&](const int8_t* in, int8_t* out, int H, int shift, int L) {
        const int tiles = 1 << (2 * shift);
        bconv_wmma<<<B * tiles, 32, 0, stream>>>(in, out, H, H, shift,
                                                 bw + L * 768, thr + L * 16);
    };
    auto pool = [&](const int8_t* in, int8_t* out, int Wo, int wo_shift) {
        const int n = B * Wo * Wo;
        pool_kernel<<<n / 256, 256, 0, stream>>>(in, out, wo_shift);
    };

    bconv(act0, act1, 64, 3, 0);
    bconv(act1, act0, 64, 3, 1);
    bconv(act0, act1, 64, 3, 2);
    pool (act1, act0, 32, 5);
    bconv(act0, act1, 32, 2, 3);
    bconv(act1, act0, 32, 2, 4);
    pool (act0, act1, 16, 4);
    bconv(act1, act0, 16, 1, 5);
    bconv(act0, act1, 16, 1, 6);
    pool (act1, act0,  8, 3);
    bconv(act0, act1,  8, 0, 7);

    head_kernel<<<B, 64, 0, stream>>>(act1, w9p, b9, g10, be10, m10, v10,
                                      (float*)d_out);
}
